// HeteroSpatialEncoder_21861383537420
// MI455X (gfx1250) — matrix operand order repaired and checked by reading, not compiled, not run on hardware
//
#include <hip/hip_runtime.h>

#define NN 50000
#define EE 800000

typedef __attribute__((ext_vector_type(16))) __bf16 v16bf;
typedef __attribute__((ext_vector_type(8)))  float  v8f;

static __device__ __forceinline__ float atomicMaxF(float* addr, float val) {
  return (val >= 0.0f)
    ? __int_as_float(atomicMax((int*)addr, __float_as_int(val)))
    : __uint_as_float(atomicMin((unsigned int*)addr, __float_as_uint(val)));
}

// ---------- small utility kernels ----------
__global__ void k_cvt_bf16(const float* __restrict__ in, __bf16* __restrict__ out, int n) {
  int i = blockIdx.x * blockDim.x + threadIdx.x;
  if (i < n) out[i] = (__bf16)in[i];
}

// W[k][n] row-major (K x Ncol) -> Wt[n][k] bf16
__global__ void k_transpose_bf16(const float* __restrict__ W, __bf16* __restrict__ Wt,
                                 int K, int Ncol) {
  int i = blockIdx.x * blockDim.x + threadIdx.x;
  if (i >= K * Ncol) return;
  int n = i / K, kk = i - n * K;
  Wt[i] = (__bf16)W[kk * Ncol + n];
}

__global__ void k_fill(float* __restrict__ p, float v, int n) {
  int i = blockIdx.x * blockDim.x + threadIdx.x;
  if (i < n) p[i] = v;
}

// ---------- Fused WMMA GEMM: {q,k,v,skip} = x_bf[N,128] @ W*[128,128] + b* ----------
// One wave per 16-wide column tile; A-fragment loaded once per k-step, reused for
// all four weight matrices -> 16 v_wmma per wave per row-strip.
__global__ void __launch_bounds__(256)
k_gemm_qkvs(const __bf16* __restrict__ A,
            const __bf16* __restrict__ Wt0, const __bf16* __restrict__ Wt1,
            const __bf16* __restrict__ Wt2, const __bf16* __restrict__ Wt3,
            const float* __restrict__ b0, const float* __restrict__ b1,
            const float* __restrict__ b2, const float* __restrict__ b3,
            float* __restrict__ oq, float* __restrict__ ok, float* __restrict__ ov,
            float* __restrict__ hcat, int hcat_off) {
  int lane = threadIdx.x & 31;
  int wave = threadIdx.x >> 5;      // 0..7 -> column tile
  int row0 = blockIdx.x * 16;
  int col0 = wave * 16;
  int half = lane >> 4;
  int l15  = lane & 15;

  const __bf16* Wts[4]  = { Wt0, Wt1, Wt2, Wt3 };
  const float*  bias[4] = { b0, b1, b2, b3 };

  v8f acc[4];
  #pragma unroll
  for (int mtx = 0; mtx < 4; ++mtx) {
    float bvv = bias[mtx][col0 + l15];
    #pragma unroll
    for (int i = 0; i < 8; ++i) acc[mtx][i] = bvv;
  }

  const __bf16* arow = A + (row0 + l15) * 128;
  int bofs = (col0 + l15) * 128;

  #pragma unroll
  for (int ks = 0; ks < 4; ++ks) {
    int k0 = ks * 32;
    union { v16bf v; uint4 u[2]; } a;
    const __bf16* pa = arow + k0 + (half ? 8 : 0);     // K: {0..7,16..23} / {8..15,24..31}
    a.u[0] = *(const uint4*)pa;
    a.u[1] = *(const uint4*)(pa + 16);
    #pragma unroll
    for (int mtx = 0; mtx < 4; ++mtx) {
      union { v16bf v; uint4 u[2]; } b;
      const __bf16* pb = Wts[mtx] + bofs + k0 + (half ? 8 : 0);  // K: {0..7,16..23} / {8..15,24..31}
      b.u[0] = *(const uint4*)pb;
      b.u[1] = *(const uint4*)(pb + 16);
      acc[mtx] = __builtin_amdgcn_wmma_f32_16x16x32_bf16(false, a.v, false, b.v,
                                                         (short)0, acc[mtx], false, false);
    }
  }
  int n = col0 + l15;
  int mbase = row0 + (half ? 8 : 0);
  #pragma unroll
  for (int i = 0; i < 8; ++i) {
    oq[(mbase + i) * 128 + n]              = acc[0][i];
    ok[(mbase + i) * 128 + n]              = acc[1][i];
    ov[(mbase + i) * 128 + n]              = acc[2][i];
    hcat[(mbase + i) * 256 + hcat_off + n] = acc[3][i];
  }
}

// ---------- WMMA GEMM: out[N,32] = relu(Hcat[N,256] @ Wlin[256,32] + b) ----------
__global__ void __launch_bounds__(64)
k_gemm_final(const float* __restrict__ Hcat, const __bf16* __restrict__ Wt,
             const float* __restrict__ bias, float* __restrict__ out) {
  int lane = threadIdx.x & 31;
  int wave = threadIdx.x >> 5;      // 0..1
  int row0 = blockIdx.x * 16;
  int col0 = wave * 16;
  int half = lane >> 4;
  int l15  = lane & 15;

  v8f acc;
  float bvv = bias[col0 + l15];
  #pragma unroll
  for (int i = 0; i < 8; ++i) acc[i] = bvv;

  const float*  arow = Hcat + (row0 + l15) * 256;
  const __bf16* brow = Wt   + (col0 + l15) * 256;

  #pragma unroll
  for (int ks = 0; ks < 8; ++ks) {
    int k0 = ks * 32;
    union { v16bf v; uint4 u[2]; } a, b;
    const float* pa = arow + k0 + (half ? 8 : 0);
    float4 f0 = *(const float4*)pa;
    float4 f1 = *(const float4*)(pa + 4);
    float4 f2 = *(const float4*)(pa + 16);
    float4 f3 = *(const float4*)(pa + 20);
    a.v[0]=(__bf16)f0.x; a.v[1]=(__bf16)f0.y; a.v[2]=(__bf16)f0.z; a.v[3]=(__bf16)f0.w;
    a.v[4]=(__bf16)f1.x; a.v[5]=(__bf16)f1.y; a.v[6]=(__bf16)f1.z; a.v[7]=(__bf16)f1.w;
    a.v[8]=(__bf16)f2.x; a.v[9]=(__bf16)f2.y; a.v[10]=(__bf16)f2.z; a.v[11]=(__bf16)f2.w;
    a.v[12]=(__bf16)f3.x; a.v[13]=(__bf16)f3.y; a.v[14]=(__bf16)f3.z; a.v[15]=(__bf16)f3.w;
    const __bf16* pb = brow + k0 + (half ? 8 : 0);
    b.u[0] = *(const uint4*)pb;
    b.u[1] = *(const uint4*)(pb + 16);
    acc = __builtin_amdgcn_wmma_f32_16x16x32_bf16(false, a.v, false, b.v,
                                                  (short)0, acc, false, false);
  }
  int n = col0 + l15;
  int mbase = row0 + (half ? 8 : 0);
  #pragma unroll
  for (int i = 0; i < 8; ++i) out[(mbase + i) * 32 + n] = fmaxf(acc[i], 0.0f);
}

// ---------- Edge pass A: alpha_raw = <q[dst], k[src]+e>/sqrt(C), atomic max ----------
// One wave per edge; lane handles 4 channels; 8 lanes per head.
// We (16x128, 8KB) staged in LDS once per block to keep VMEM free for gathers.
__global__ void __launch_bounds__(256)
k_edge_a(const int* __restrict__ ei, const float* __restrict__ ea,
         const float* __restrict__ We, const float* __restrict__ q,
         const float* __restrict__ k, float* __restrict__ alpha,
         float* __restrict__ m) {
  __shared__ float sWe[16 * 128];
  for (int i = threadIdx.x; i < 16 * 128; i += 256) sWe[i] = We[i];
  __syncthreads();

  int e = (blockIdx.x * blockDim.x + threadIdx.x) >> 5;
  int lane = threadIdx.x & 31;
  if (e >= EE) return;
  int src = ei[e], dst = ei[EE + e];
  float eav = ea[e * 16 + (lane & 15)];
  int c0 = lane * 4;
  float4 qv = *(const float4*)(q + dst * 128 + c0);
  float4 kv = *(const float4*)(k + src * 128 + c0);
  float feat0 = 0.f, feat1 = 0.f, feat2 = 0.f, feat3 = 0.f;
  #pragma unroll
  for (int j = 0; j < 16; ++j) {
    float a = __shfl(eav, j, 32);
    float4 wv = *(const float4*)(sWe + j * 128 + c0);
    feat0 += a * wv.x; feat1 += a * wv.y; feat2 += a * wv.z; feat3 += a * wv.w;
  }
  float part = qv.x * (kv.x + feat0) + qv.y * (kv.y + feat1)
             + qv.z * (kv.z + feat2) + qv.w * (kv.w + feat3);
  part += __shfl_xor(part, 1, 32);
  part += __shfl_xor(part, 2, 32);
  part += __shfl_xor(part, 4, 32);
  if ((lane & 7) == 0) {
    int h = lane >> 3;
    float al = part * 0.17677669529663687f;  // 1/sqrt(32)
    alpha[e * 4 + h] = al;
    atomicMaxF(&m[dst * 4 + h], al);
  }
}

// ---------- Edge pass B: ex = exp(a - m[dst]); atomic segment sum ----------
__global__ void __launch_bounds__(256)
k_edge_b(const int* __restrict__ ei, float* __restrict__ alpha,
         const float* __restrict__ m, float* __restrict__ s) {
  int t = blockIdx.x * blockDim.x + threadIdx.x;
  if (t >= EE * 4) return;
  int e = t >> 2, h = t & 3;
  int dst = ei[EE + e];
  float ex = __expf(alpha[t] - m[dst * 4 + h]);
  alpha[t] = ex;
  atomicAdd(&s[dst * 4 + h], ex);
}

// ---------- Edge pass C: scatter coef*(v[src]+e) into hcat[dst] ----------
__global__ void __launch_bounds__(256)
k_edge_c(const int* __restrict__ ei, const float* __restrict__ ea,
         const float* __restrict__ We, const float* __restrict__ v,
         const float* __restrict__ alpha, const float* __restrict__ s,
         float* __restrict__ hcat, int col_off) {
  __shared__ float sWe[16 * 128];
  for (int i = threadIdx.x; i < 16 * 128; i += 256) sWe[i] = We[i];
  __syncthreads();

  int e = (blockIdx.x * blockDim.x + threadIdx.x) >> 5;
  int lane = threadIdx.x & 31;
  if (e >= EE) return;
  int src = ei[e], dst = ei[EE + e];
  int h = lane >> 3;
  float coef = alpha[e * 4 + h] / (s[dst * 4 + h] + 1e-16f);
  float eav = ea[e * 16 + (lane & 15)];
  int c0 = lane * 4;
  float4 vv = *(const float4*)(v + src * 128 + c0);
  float feat0 = 0.f, feat1 = 0.f, feat2 = 0.f, feat3 = 0.f;
  #pragma unroll
  for (int j = 0; j < 16; ++j) {
    float a = __shfl(eav, j, 32);
    float4 wv = *(const float4*)(sWe + j * 128 + c0);
    feat0 += a * wv.x; feat1 += a * wv.y; feat2 += a * wv.z; feat3 += a * wv.w;
  }
  float* o = hcat + dst * 256 + col_off + c0;
  atomicAdd(o + 0, coef * (vv.x + feat0));
  atomicAdd(o + 1, coef * (vv.y + feat1));
  atomicAdd(o + 2, coef * (vv.z + feat2));
  atomicAdd(o + 3, coef * (vv.w + feat3));
}

extern "C" void kernel_launch(void* const* d_in, const int* in_sizes, int n_in,
                              void* d_out, int out_size, void* d_ws, size_t ws_size,
                              hipStream_t stream) {
  const float* x     = (const float*)d_in[0];
  const int*   ei[2] = { (const int*)d_in[1], (const int*)d_in[3] };
  const float* ea[2] = { (const float*)d_in[2], (const float*)d_in[4] };
  const float* Wq[2] = { (const float*)d_in[5],  (const float*)d_in[14] };
  const float* bq[2] = { (const float*)d_in[6],  (const float*)d_in[15] };
  const float* Wk[2] = { (const float*)d_in[7],  (const float*)d_in[16] };
  const float* bk[2] = { (const float*)d_in[8],  (const float*)d_in[17] };
  const float* Wv[2] = { (const float*)d_in[9],  (const float*)d_in[18] };
  const float* bv[2] = { (const float*)d_in[10], (const float*)d_in[19] };
  const float* We[2] = { (const float*)d_in[11], (const float*)d_in[20] };
  const float* Ws[2] = { (const float*)d_in[12], (const float*)d_in[21] };
  const float* bs[2] = { (const float*)d_in[13], (const float*)d_in[22] };
  const float* Wlin = (const float*)d_in[23];
  const float* blin = (const float*)d_in[24];

  char* w = (char*)d_ws;
  auto alloc = [&](size_t bytes) -> char* {
    char* p = w; w += (bytes + 255) & ~(size_t)255; return p;
  };
  __bf16* x_bf = (__bf16*)alloc((size_t)NN * 128 * 2);
  __bf16* Wt[2][4];
  const float* Wsrc[2][4] = { { Wq[0], Wk[0], Wv[0], Ws[0] },
                              { Wq[1], Wk[1], Wv[1], Ws[1] } };
  for (int r = 0; r < 2; ++r)
    for (int j = 0; j < 4; ++j)
      Wt[r][j] = (__bf16*)alloc(128 * 128 * 2);
  __bf16* Wlin_t = (__bf16*)alloc(32 * 256 * 2);
  float* q[2];  float* kk[2];  float* vv[2];
  for (int r = 0; r < 2; ++r) {
    q[r]  = (float*)alloc((size_t)NN * 128 * 4);
    kk[r] = (float*)alloc((size_t)NN * 128 * 4);
    vv[r] = (float*)alloc((size_t)NN * 128 * 4);
  }
  float* hcat = (float*)alloc((size_t)NN * 256 * 4);
  float* alp[2] = { (float*)alloc((size_t)EE * 4 * 4), (float*)alloc((size_t)EE * 4 * 4) };
  float* mbuf = (float*)alloc((size_t)2 * NN * 4 * 4);
  float* sbuf = (float*)alloc((size_t)2 * NN * 4 * 4);

  // precompute: bf16 x, transposed bf16 weights, softmax state init
  k_cvt_bf16<<<(NN * 128 + 255) / 256, 256, 0, stream>>>(x, x_bf, NN * 128);
  for (int r = 0; r < 2; ++r)
    for (int j = 0; j < 4; ++j)
      k_transpose_bf16<<<(128 * 128 + 255) / 256, 256, 0, stream>>>(Wsrc[r][j], Wt[r][j], 128, 128);
  k_transpose_bf16<<<(256 * 32 + 255) / 256, 256, 0, stream>>>(Wlin, Wlin_t, 256, 32);
  k_fill<<<(2 * NN * 4 + 255) / 256, 256, 0, stream>>>(mbuf, -__builtin_huge_valf(), 2 * NN * 4);
  k_fill<<<(2 * NN * 4 + 255) / 256, 256, 0, stream>>>(sbuf, 0.0f, 2 * NN * 4);

  const int gemm_grid = NN / 16;     // 3125
  const int edge_grid = EE / 8;      // 8 waves/block

  for (int r = 0; r < 2; ++r) {
    float* mr = mbuf + (size_t)r * NN * 4;
    float* sr = sbuf + (size_t)r * NN * 4;
    // fused q,k,v,skip projections: A-fragment reused for 4 WMMAs per k-step
    k_gemm_qkvs<<<gemm_grid, 256, 0, stream>>>(
        x_bf, Wt[r][0], Wt[r][1], Wt[r][2], Wt[r][3],
        bq[r], bk[r], bv[r], bs[r],
        q[r], kk[r], vv[r], hcat, r * 128);
    // segment-softmax attention + scatter
    k_edge_a<<<edge_grid, 256, 0, stream>>>(ei[r], ea[r], We[r], q[r], kk[r], alp[r], mr);
    k_edge_b<<<(EE * 4 + 255) / 256, 256, 0, stream>>>(ei[r], alp[r], mr, sr);
    k_edge_c<<<edge_grid, 256, 0, stream>>>(ei[r], ea[r], We[r], vv[r], alp[r], sr, hcat, r * 128);
  }

  // fused output: relu(hcat @ Wlin + b)
  k_gemm_final<<<gemm_grid, 64, 0, stream>>>(hcat, Wlin_t, blin, (float*)d_out);
}
